// HYV3DecoderLayer_90099823935491
// MI455X (gfx1250) — matrix operand order repaired and checked by reading, not compiled, not run on hardware
//
#include <hip/hip_runtime.h>

// ---------------------------------------------------------------------------
// MoE decoder layer for MI455X (gfx1250, wave32, WMMA bf16 16x16x32).
// - All GEMMs + both attention matmuls use v_wmma_f32_16x16x32_bf16.
// - Fragment loads are contiguous 2x ds_load_b128 (A row-major, B transposed).
// - bf16 global->LDS tile copies use GLOBAL_LOAD_ASYNC_TO_LDS_B128 (ASYNCcnt).
// - fp32 weight tiles converted with packed bf16 dword stores (k-pairs).
// - Routed experts computed sparsely (top-4/16) via token compaction.
// ---------------------------------------------------------------------------

#define DEV static __device__ __forceinline__

typedef __attribute__((ext_vector_type(16))) __bf16 v16bf;
typedef __attribute__((ext_vector_type(8)))  float  v8f;

constexpr int   T_ = 2048, H_ = 2048, NH_ = 16, NKV_ = 8, D_ = 128;
constexpr int   E_ = 16, KTOP = 4, I_ = 1024, SI_ = 1024;
constexpr int   QKVN = (NH_ + 2 * NKV_) * D_;           // 4096
constexpr float EPSF = 1e-5f;
constexpr float THETAF = 10000.0f;

// -------------------------- WMMA helpers -----------------------------------

DEV v8f wmma_bf16(v16bf a, v16bf b, v8f c) {
  return __builtin_amdgcn_wmma_f32_16x16x32_bf16(
      /*neg_a=*/false, a, /*neg_b=*/false, b,
      /*c_mod=*/(short)0, c, /*reuse_a=*/false, /*reuse_b=*/false);
}

// A fragment: 16x32 bf16 from row-major LDS tile (pitch ld, ld%8==0).
// lane<16: M=lane, K=0..15; lane>=16: M=lane-16, K=16..31.
DEV v16bf load_frag_a(const __bf16* S, int ld, int row0, int k0, int lane) {
  int half = lane >> 4, m = lane & 15;
  const __bf16* p = S + (size_t)(row0 + m) * ld + k0 + half * 16;
  v16bf a;
  *((uint4*)&a)     = *(const uint4*)p;        // K = k0+half*16 .. +7
  *((uint4*)&a + 1) = *(const uint4*)(p + 8);  // K = k0+half*16+8 .. +7
  return a;
}

// B fragment: 32x16 bf16 where LDS holds B transposed: S[n*ld + k].
// lane<16: N=lane, K=k0..k0+15 ; lane>=16: N=lane-16, K=k0+16..k0+31.
DEV v16bf load_frag_bT(const __bf16* S, int ld, int k0, int col0, int lane) {
  int n = lane & 15, kb = (lane >> 4) * 16;
  const __bf16* p = S + (size_t)(col0 + n) * ld + k0 + kb;
  v16bf b;
  *((uint4*)&b)     = *(const uint4*)p;
  *((uint4*)&b + 1) = *(const uint4*)(p + 8);
  return b;
}

// C/D element mapping: VGPR r -> M = r + 8*(lane>=16), N = lane&15.

// ---------------- CDNA5 async global->LDS copy (ASYNCcnt) -------------------

// Copy 16 bytes per lane from global to LDS via the async DMA path.
DEV void async_copy16(__bf16* lds_dst, const __bf16* gsrc) {
  unsigned ldsa = (unsigned)(size_t)lds_dst;
  asm volatile("global_load_async_to_lds_b128 %0, %1, off"
               :: "v"(ldsa), "v"(gsrc) : "memory");
}
DEV void async_wait0() {
  asm volatile("s_wait_asynccnt 0x0" ::: "memory");
}

// Pack two floats into one packed-bf16 dword and store to LDS (k-pair store).
DEV void store_bf16_pair(__bf16* dst, float lo, float hi) {
  union { unsigned u; __bf16 h[2]; } pk;
  pk.h[0] = (__bf16)lo;
  pk.h[1] = (__bf16)hi;
  *(unsigned*)dst = pk.u;
}

// ------------------------ elementwise kernels -------------------------------

__global__ __launch_bounds__(256) void rmsnorm_to_bf16_kernel(
    const float* __restrict__ x, const float* __restrict__ w,
    __bf16* __restrict__ out, int Hd) {
  int t = blockIdx.x;
  __shared__ float red[256];
  float s = 0.f;
  for (int i = threadIdx.x; i < Hd; i += 256) {
    float v = x[(size_t)t * Hd + i];
    s += v * v;
  }
  red[threadIdx.x] = s;
  __syncthreads();
  for (int o = 128; o > 0; o >>= 1) {
    if (threadIdx.x < o) red[threadIdx.x] += red[threadIdx.x + o];
    __syncthreads();
  }
  float r = rsqrtf(red[0] / (float)Hd + EPSF);
  for (int i = threadIdx.x; i < Hd; i += 256)
    out[(size_t)t * Hd + i] = (__bf16)(x[(size_t)t * Hd + i] * r * w[i]);
}

// QK rmsnorm + RoPE + bf16 pack of q/k/v. blockDim = 128 (= D).
__global__ __launch_bounds__(128) void qk_rope_kernel(
    const float* __restrict__ qkv, const int* __restrict__ positions,
    const float* __restrict__ qw, const float* __restrict__ kw,
    __bf16* __restrict__ qb, __bf16* __restrict__ kb, __bf16* __restrict__ vb) {
  int t = blockIdx.x;
  int tid = threadIdx.x;
  float pos = (float)positions[t];
  __shared__ float buf[128];
  __shared__ float red[128];
  const float* row = qkv + (size_t)t * QKVN;

  int dd = tid & 63;
  float inv = __expf(-(2.0f * (float)dd / (float)D_) * logf(THETAF));
  float ang = pos * inv;
  float c = __cosf(ang), s = __sinf(ang);

  for (int hh = 0; hh < NH_ + NKV_; ++hh) {
    bool isq = hh < NH_;
    const float* src = row + (isq ? hh * D_ : NH_ * D_ + (hh - NH_) * D_);
    float v = src[tid];
    red[tid] = v * v;
    __syncthreads();
    for (int o = 64; o > 0; o >>= 1) {
      if (tid < o) red[tid] += red[tid + o];
      __syncthreads();
    }
    float r = rsqrtf(red[0] / (float)D_ + EPSF);
    float nv = v * r * (isq ? qw[tid] : kw[tid]);
    buf[tid] = nv;
    __syncthreads();
    float x1 = buf[dd], x2 = buf[dd + 64];
    float outv = (tid < 64) ? (x1 * c - x2 * s) : (x2 * c + x1 * s);
    if (isq)
      qb[((size_t)t * NH_ + hh) * D_ + tid] = (__bf16)outv;
    else
      kb[((size_t)t * NKV_ + (hh - NH_)) * D_ + tid] = (__bf16)outv;
    __syncthreads();
  }
  for (int hh = 0; hh < NKV_; ++hh) {
    float v = row[NH_ * D_ + NKV_ * D_ + hh * D_ + tid];
    vb[((size_t)t * NKV_ + hh) * D_ + tid] = (__bf16)v;
  }
}

// -------------------------- generic WMMA GEMM -------------------------------
// C[M,N](f32) = A[M,K](bf16) @ B[K,N](f32->bf16).
// MODE: 0 C=acc, 1 C=SRC+acc (residual), 2 C+=acc (accumulate).
template <int MODE>
__global__ __launch_bounds__(256) void gemm_bf16_kernel(
    const __bf16* __restrict__ A, const float* __restrict__ Bw,
    float* __restrict__ C, const float* __restrict__ SRC,
    int M, int N, int Kd) {
  constexpr int BM = 128, BN = 128, BK = 32;
  constexpr int LDA = BK + 8, LDT = BK + 8;      // 40 (80B pitch, 16B aligned)
  __shared__ __bf16 As[BM * LDA];
  __shared__ __bf16 BsT[BN * LDT];               // transposed: [n][k]
  int m0 = blockIdx.x * BM, n0 = blockIdx.y * BN;
  int tid = threadIdx.x, lane = tid & 31, wid = tid >> 5;
  int wm = (wid & 3) * 32, wn = (wid >> 2) * 64;

  v8f zero = {};
  v8f acc[2][4];
#pragma unroll
  for (int a = 0; a < 2; ++a)
#pragma unroll
    for (int b = 0; b < 4; ++b) acc[a][b] = zero;

  for (int k0 = 0; k0 < Kd; k0 += BK) {
    {
      // A tile: bf16 global -> LDS via async DMA (2x 16B per thread)
      int r = tid >> 1, cb = (tid & 1) * 16;
      const __bf16* src = A + (size_t)(m0 + r) * Kd + k0 + cb;
      __bf16* dst = &As[r * LDA + cb];
      async_copy16(dst, src);
      async_copy16(dst + 8, src + 8);
    }
    {
      // B tile: fp32 -> packed bf16 k-pairs into transposed LDS tile
      int kp = tid & 15;            // k pair: rows 2kp, 2kp+1
      int nb = (tid >> 4) * 8;      // 8 columns
      const float* r0 = Bw + (size_t)(k0 + 2 * kp) * N + n0 + nb;
      const float* r1 = r0 + N;
      if (k0 + BK < Kd)
        __builtin_prefetch(r0 + (size_t)BK * N, 0, 1);
      float4 f0[2], f1[2];
      f0[0] = ((const float4*)r0)[0]; f0[1] = ((const float4*)r0)[1];
      f1[0] = ((const float4*)r1)[0]; f1[1] = ((const float4*)r1)[1];
      const float* a0 = (const float*)f0;
      const float* a1 = (const float*)f1;
#pragma unroll
      for (int j = 0; j < 8; ++j)
        store_bf16_pair(&BsT[(nb + j) * LDT + 2 * kp], a0[j], a1[j]);
    }
    async_wait0();
    __syncthreads();
#pragma unroll
    for (int mi = 0; mi < 2; ++mi) {
      v16bf a = load_frag_a(As, LDA, wm + mi * 16, 0, lane);
#pragma unroll
      for (int ni = 0; ni < 4; ++ni) {
        v16bf b = load_frag_bT(BsT, LDT, 0, wn + ni * 16, lane);
        acc[mi][ni] = wmma_bf16(a, b, acc[mi][ni]);
      }
    }
    __syncthreads();
  }
#pragma unroll
  for (int mi = 0; mi < 2; ++mi)
#pragma unroll
    for (int ni = 0; ni < 4; ++ni)
#pragma unroll
      for (int r = 0; r < 8; ++r) {
        int m = m0 + wm + mi * 16 + r + ((lane >> 4) << 3);
        int n = n0 + wn + ni * 16 + (lane & 15);
        size_t idx = (size_t)m * N + n;
        float v = acc[mi][ni][r];
        if (MODE == 1) v += SRC[idx];
        if (MODE == 2) v += C[idx];
        C[idx] = v;
      }
}

// ------------------- GLU GEMM: act = silu(x@Wg) * (x@Wu) --------------------
// Bw is [Kd, 2*Icols] (g cols then u cols); outputs bf16 act [rows, Icols].
// GATHER: A rows from tok_list[e*T + r], out row = base[e]+r, M = counts[e].
template <bool GATHER>
__global__ __launch_bounds__(256) void glu_gemm_kernel(
    const __bf16* __restrict__ A, const float* __restrict__ Bw,
    __bf16* __restrict__ ACT, int Kd, int Icols,
    const int* __restrict__ tok_list, const int* __restrict__ counts,
    const int* __restrict__ basev, int Trows) {
  constexpr int BM = 128, BN = 64, BK = 32;
  constexpr int LDA = BK + 8, LDT = BK + 8;
  int e = GATHER ? blockIdx.z : 0;
  int Me = GATHER ? counts[e] : Trows;
  int m0 = blockIdx.x * BM;
  if (m0 >= Me) return;
  int n0 = blockIdx.y * BN;
  const float* Bg = Bw + (size_t)e * Kd * (2 * Icols);

  __shared__ __bf16 As[BM * LDA];
  __shared__ __bf16 BgT[BN * LDT];
  __shared__ __bf16 BuT[BN * LDT];
  int tid = threadIdx.x, lane = tid & 31, wid = tid >> 5;
  int wm = wid * 16;

  v8f zero = {};
  v8f ag[4], au[4];
#pragma unroll
  for (int i = 0; i < 4; ++i) { ag[i] = zero; au[i] = zero; }

  for (int k0 = 0; k0 < Kd; k0 += BK) {
    {
      int r = tid >> 1, cb = (tid & 1) * 16;
      int row = m0 + r;
      int lr = row < Me ? row : Me - 1;
      int arow = GATHER ? tok_list[e * Trows + lr] : lr;
      const __bf16* src = A + (size_t)arow * Kd + k0 + cb;
      __bf16* dst = &As[r * LDA + cb];
      async_copy16(dst, src);
      async_copy16(dst + 8, src + 8);
    }
    {
      int kp = tid & 15;            // k pair: rows 2kp, 2kp+1
      int nb = (tid >> 4) * 4;      // 4 columns
      const float* g0 = Bg + (size_t)(k0 + 2 * kp) * (2 * Icols) + n0 + nb;
      const float* g1 = g0 + 2 * Icols;
      const float* u0 = g0 + Icols;
      const float* u1 = g1 + Icols;
      if (k0 + BK < Kd)
        __builtin_prefetch(g0 + (size_t)BK * (2 * Icols), 0, 1);
      float4 fg0 = *(const float4*)g0;
      float4 fg1 = *(const float4*)g1;
      float4 fu0 = *(const float4*)u0;
      float4 fu1 = *(const float4*)u1;
      const float* pg0 = (const float*)&fg0;
      const float* pg1 = (const float*)&fg1;
      const float* pu0 = (const float*)&fu0;
      const float* pu1 = (const float*)&fu1;
#pragma unroll
      for (int j = 0; j < 4; ++j) {
        store_bf16_pair(&BgT[(nb + j) * LDT + 2 * kp], pg0[j], pg1[j]);
        store_bf16_pair(&BuT[(nb + j) * LDT + 2 * kp], pu0[j], pu1[j]);
      }
    }
    async_wait0();
    __syncthreads();
    v16bf a = load_frag_a(As, LDA, wm, 0, lane);
#pragma unroll
    for (int ni = 0; ni < 4; ++ni) {
      v16bf bg = load_frag_bT(BgT, LDT, 0, ni * 16, lane);
      ag[ni] = wmma_bf16(a, bg, ag[ni]);
      v16bf bu = load_frag_bT(BuT, LDT, 0, ni * 16, lane);
      au[ni] = wmma_bf16(a, bu, au[ni]);
    }
    __syncthreads();
  }
#pragma unroll
  for (int ni = 0; ni < 4; ++ni)
#pragma unroll
    for (int r = 0; r < 8; ++r) {
      int ml = wm + r + ((lane >> 4) << 3);
      int m = m0 + ml;
      if (m >= Me) continue;
      int col = n0 + ni * 16 + (lane & 15);
      float g = ag[ni][r], u = au[ni][r];
      float act = g / (1.f + __expf(-g)) * u;
      int orow = GATHER ? (basev[e] + m) : m;
      ACT[(size_t)orow * Icols + col] = (__bf16)act;
    }
}

// ------------- expert down GEMM over compacted rows (plain store) -----------
__global__ __launch_bounds__(256) void expert_down_gemm_kernel(
    const __bf16* __restrict__ ACT, const float* __restrict__ Dw,
    float* __restrict__ OUT, const int* __restrict__ counts,
    const int* __restrict__ basev) {
  constexpr int BM = 128, BN = 128, BK = 32;
  constexpr int LDA = BK + 8, LDT = BK + 8;
  int e = blockIdx.z;
  int Me = counts[e];
  int m0 = blockIdx.x * BM;
  if (m0 >= Me) return;
  int gb = basev[e];
  const __bf16* A = ACT + (size_t)gb * I_;
  const float* Bw = Dw + (size_t)e * I_ * H_;
  float* C = OUT + (size_t)gb * H_;
  int n0 = blockIdx.y * BN;

  __shared__ __bf16 As[BM * LDA];
  __shared__ __bf16 BsT[BN * LDT];
  int tid = threadIdx.x, lane = tid & 31, wid = tid >> 5;
  int wm = (wid & 3) * 32, wn = (wid >> 2) * 64;

  v8f zero = {};
  v8f acc[2][4];
#pragma unroll
  for (int a = 0; a < 2; ++a)
#pragma unroll
    for (int b = 0; b < 4; ++b) acc[a][b] = zero;

  for (int k0 = 0; k0 < I_; k0 += BK) {
    {
      int r = tid >> 1, cb = (tid & 1) * 16;
      int lr = (m0 + r) < Me ? (m0 + r) : (Me - 1);
      const __bf16* src = A + (size_t)lr * I_ + k0 + cb;
      __bf16* dst = &As[r * LDA + cb];
      async_copy16(dst, src);
      async_copy16(dst + 8, src + 8);
    }
    {
      int kp = tid & 15;
      int nb = (tid >> 4) * 8;
      const float* r0 = Bw + (size_t)(k0 + 2 * kp) * H_ + n0 + nb;
      const float* r1 = r0 + H_;
      if (k0 + BK < I_)
        __builtin_prefetch(r0 + (size_t)BK * H_, 0, 1);
      float4 f0[2], f1[2];
      f0[0] = ((const float4*)r0)[0]; f0[1] = ((const float4*)r0)[1];
      f1[0] = ((const float4*)r1)[0]; f1[1] = ((const float4*)r1)[1];
      const float* a0 = (const float*)f0;
      const float* a1 = (const float*)f1;
#pragma unroll
      for (int j = 0; j < 8; ++j)
        store_bf16_pair(&BsT[(nb + j) * LDT + 2 * kp], a0[j], a1[j]);
    }
    async_wait0();
    __syncthreads();
#pragma unroll
    for (int mi = 0; mi < 2; ++mi) {
      v16bf a = load_frag_a(As, LDA, wm + mi * 16, 0, lane);
#pragma unroll
      for (int ni = 0; ni < 4; ++ni) {
        v16bf b = load_frag_bT(BsT, LDT, 0, wn + ni * 16, lane);
        acc[mi][ni] = wmma_bf16(a, b, acc[mi][ni]);
      }
    }
    __syncthreads();
  }
#pragma unroll
  for (int mi = 0; mi < 2; ++mi)
#pragma unroll
    for (int ni = 0; ni < 4; ++ni)
#pragma unroll
      for (int r = 0; r < 8; ++r) {
        int m = m0 + wm + mi * 16 + r + ((lane >> 4) << 3);
        if (m >= Me) continue;
        int n = n0 + wn + ni * 16 + (lane & 15);
        C[(size_t)m * H_ + n] = acc[mi][ni][r];
      }
}

// ------------------------- flash attention (WMMA) ---------------------------
constexpr int LDF = 132;   // f32 score pitch
constexpr int LDB = 136;   // bf16 tile pitch (272B, 16B aligned)
constexpr size_t ATT_LDS =
    (size_t)128 * LDF * 4 + 4 * (size_t)128 * LDB * 2 + 3 * 128 * 4; // ~208 KB

__global__ __launch_bounds__(256) void attention_kernel(
    const __bf16* __restrict__ qb, const __bf16* __restrict__ kb,
    const __bf16* __restrict__ vb, __bf16* __restrict__ attnb) {
  extern __shared__ char smem[];
  float* Sf = (float*)smem;
  __bf16* Qs  = (__bf16*)(smem + (size_t)128 * LDF * 4); // [qrow][d]
  __bf16* Ks  = Qs + 128 * LDB;                          // [key][d]
  __bf16* VsT = Ks + 128 * LDB;                          // [d][key] transposed
  __bf16* Ps  = VsT + 128 * LDB;                         // [qrow][key]
  float* row_max = (float*)(Ps + 128 * LDB);
  float* row_sum = row_max + 128;
  float* alph = row_sum + 128;

  const int h = blockIdx.x;
  const int qt = blockIdx.y;
  const int hk = h / (NH_ / NKV_);
  const int tid = threadIdx.x, lane = tid & 31, wid = tid >> 5;
  const int wm = (wid & 3) * 32, wn = (wid >> 2) * 64;
  const float scale = 0.08838834764831845f; // 1/sqrt(D)

  {
    int r = tid >> 1, db = (tid & 1) * 64;
    const __bf16* src = qb + ((size_t)(qt * 128 + r) * NH_ + h) * D_ + db;
    __bf16* dst = Qs + r * LDB + db;
#pragma unroll
    for (int j = 0; j < 8; ++j) async_copy16(dst + j * 8, src + j * 8);
  }
  if (tid < 128) { row_max[tid] = -1e30f; row_sum[tid] = 0.f; }

  v8f zero = {};
  v8f acc_o[2][4];
#pragma unroll
  for (int a = 0; a < 2; ++a)
#pragma unroll
    for (int b = 0; b < 4; ++b) acc_o[a][b] = zero;
  async_wait0();
  __syncthreads();

  for (int kt = 0; kt <= qt; ++kt) {
    {
      // K tile: async copy row-major
      int r = tid >> 1, db = (tid & 1) * 64;
      const __bf16* ks = kb + ((size_t)(kt * 128 + r) * NKV_ + hk) * D_ + db;
      __bf16* kd = Ks + r * LDB + db;
#pragma unroll
      for (int j = 0; j < 8; ++j) async_copy16(kd + j * 8, ks + j * 8);
    }
    {
      // V tile transposed: key-pairs packed into dword stores
      int kp = tid & 63, db = (tid >> 6) * 32;
      const __bf16* v0 = vb + ((size_t)(kt * 128 + 2 * kp) * NKV_ + hk) * D_ + db;
      const __bf16* v1 = vb + ((size_t)(kt * 128 + 2 * kp + 1) * NKV_ + hk) * D_ + db;
      uint4 t0[4], t1[4];
#pragma unroll
      for (int q = 0; q < 4; ++q) {
        t0[q] = ((const uint4*)v0)[q];
        t1[q] = ((const uint4*)v1)[q];
      }
      const __bf16* p0 = (const __bf16*)t0;
      const __bf16* p1 = (const __bf16*)t1;
#pragma unroll
      for (int j = 0; j < 32; ++j) {
        union { unsigned u; __bf16 h[2]; } pk;
        pk.h[0] = p0[j];
        pk.h[1] = p1[j];
        *(unsigned*)&VsT[(db + j) * LDB + 2 * kp] = pk.u;
      }
    }
    async_wait0();
    __syncthreads();

    // S = Q @ K^T   (B-fragment reads contiguous rows of Ks)
    v8f accs[2][4];
#pragma unroll
    for (int a = 0; a < 2; ++a)
#pragma unroll
      for (int b = 0; b < 4; ++b) accs[a][b] = zero;
    for (int kc = 0; kc < D_; kc += 32) {
#pragma unroll
      for (int mi = 0; mi < 2; ++mi) {
        v16bf a = load_frag_a(Qs, LDB, wm + mi * 16, kc, lane);
#pragma unroll
        for (int ni = 0; ni < 4; ++ni) {
          v16bf b = load_frag_bT(Ks, LDB, kc, wn + ni * 16, lane);
          accs[mi][ni] = wmma_bf16(a, b, accs[mi][ni]);
        }
      }
    }
#pragma unroll
    for (int mi = 0; mi < 2; ++mi)
#pragma unroll
      for (int ni = 0; ni < 4; ++ni)
#pragma unroll
        for (int r = 0; r < 8; ++r) {
          int m = wm + mi * 16 + r + ((lane >> 4) << 3);
          int c = wn + ni * 16 + (lane & 15);
          Sf[m * LDF + c] = accs[mi][ni][r];
        }
    __syncthreads();

    // online softmax: 2 threads per row
    {
      int row = tid >> 1, half = tid & 1;
      float tmax = -1e30f;
      for (int j = 0; j < 64; ++j) {
        int c = half * 64 + j;
        float s = Sf[row * LDF + c] * scale;
        if (kt == qt && c > row) s = -1e30f;
        Sf[row * LDF + c] = s;
        tmax = fmaxf(tmax, s);
      }
      tmax = fmaxf(tmax, __shfl_xor(tmax, 1, 32));
      float om = row_max[row];
      float nm = fmaxf(om, tmax);
      float al = __expf(om - nm);
      float ps = 0.f;
      for (int j = 0; j < 64; ++j) {
        int c = half * 64 + j;
        float p = __expf(Sf[row * LDF + c] - nm);
        Ps[row * LDB + c] = (__bf16)p;
        ps += p;
      }
      ps += __shfl_xor(ps, 1, 32);
      if (half == 0) {
        row_max[row] = nm;
        row_sum[row] = row_sum[row] * al + ps;
        alph[row] = al;
      }
    }
    __syncthreads();

    // O = O*alpha + P @ V   (B-fragment reads contiguous rows of VsT)
#pragma unroll
    for (int mi = 0; mi < 2; ++mi)
#pragma unroll
      for (int r = 0; r < 8; ++r) {
        int m = wm + mi * 16 + r + ((lane >> 4) << 3);
        float al = alph[m];
#pragma unroll
        for (int ni = 0; ni < 4; ++ni) acc_o[mi][ni][r] *= al;
      }
    for (int kc = 0; kc < 128; kc += 32) {
#pragma unroll
      for (int mi = 0; mi < 2; ++mi) {
        v16bf a = load_frag_a(Ps, LDB, wm + mi * 16, kc, lane);
#pragma unroll
        for (int ni = 0; ni < 4; ++ni) {
          v16bf b = load_frag_bT(VsT, LDB, kc, wn + ni * 16, lane);
          acc_o[mi][ni] = wmma_bf16(a, b, acc_o[mi][ni]);
        }
      }
    }
    __syncthreads();
  }

#pragma unroll
  for (int mi = 0; mi < 2; ++mi)
#pragma unroll
    for (int ni = 0; ni < 4; ++ni)
#pragma unroll
      for (int r = 0; r < 8; ++r) {
        int m = wm + mi * 16 + r + ((lane >> 4) << 3);
        int c = wn + ni * 16 + (lane & 15);
        float v = acc_o[mi][ni][r] / row_sum[m];
        attnb[(size_t)(qt * 128 + m) * (NH_ * D_) + h * D_ + c] = (__bf16)v;
      }
}

// ------------------------------ routing -------------------------------------

__global__ __launch_bounds__(256) void router_logits_kernel(
    const __bf16* __restrict__ x2b, const float* __restrict__ gate_w,
    float* __restrict__ logits) {
  int t = blockIdx.x * 16 + (threadIdx.x >> 4);
  int e = threadIdx.x & 15;
  float s = 0.f;
  for (int i = 0; i < H_; ++i)
    s += (float)x2b[(size_t)t * H_ + i] * gate_w[(size_t)i * E_ + e];
  logits[(size_t)t * E_ + e] = s;
}

__global__ __launch_bounds__(64) void zero_counts_kernel(int* counts) {
  if (threadIdx.x < E_) counts[threadIdx.x] = 0;
}

__global__ __launch_bounds__(256) void topk_route_kernel(
    const float* __restrict__ logits, const float* __restrict__ expert_bias,
    int* __restrict__ counts, int* __restrict__ sel_e, int* __restrict__ sel_pos,
    float* __restrict__ sel_w, int* __restrict__ tok_list) {
  int t = blockIdx.x * 256 + threadIdx.x;
  if (t >= T_) return;
  float sig[E_], biased[E_];
#pragma unroll
  for (int e = 0; e < E_; ++e) {
    float l = logits[(size_t)t * E_ + e];
    float sg = 1.f / (1.f + __expf(-l));
    sig[e] = sg;
    biased[e] = sg + expert_bias[e];
  }
  bool taken[E_];
#pragma unroll
  for (int e = 0; e < E_; ++e) taken[e] = false;
  int idx[KTOP];
  float wsum = 0.f;
#pragma unroll
  for (int k = 0; k < KTOP; ++k) {
    int best = 0;
    float bv = -1e30f;
#pragma unroll
    for (int e = 0; e < E_; ++e)
      if (!taken[e] && biased[e] > bv) { bv = biased[e]; best = e; }
    taken[best] = true;
    idx[k] = best;
    wsum += sig[best];
  }
#pragma unroll
  for (int k = 0; k < KTOP; ++k) {
    int e = idx[k];
    int pos = atomicAdd(&counts[e], 1);
    tok_list[e * T_ + pos] = t;
    sel_e[t * KTOP + k] = e;
    sel_pos[t * KTOP + k] = pos;
    sel_w[t * KTOP + k] = sig[e] / wsum; // RSF == 1.0
  }
}

__global__ void prefix_kernel(const int* __restrict__ counts, int* __restrict__ basev) {
  if (threadIdx.x == 0) {
    int s = 0;
    for (int e = 0; e < E_; ++e) { basev[e] = s; s += counts[e]; }
    basev[E_] = s;
  }
}

// out[t] += sum_k w_k * down_out[row_k]   (deterministic, no atomics)
__global__ __launch_bounds__(256) void combine_kernel(
    const int* __restrict__ sel_e, const int* __restrict__ sel_pos,
    const float* __restrict__ sel_w, const int* __restrict__ basev,
    const float* __restrict__ down_out, float* __restrict__ out) {
  int t = blockIdx.x;
  int rows[KTOP];
  float w[KTOP];
#pragma unroll
  for (int k = 0; k < KTOP; ++k) {
    rows[k] = basev[sel_e[t * KTOP + k]] + sel_pos[t * KTOP + k];
    w[k] = sel_w[t * KTOP + k];
  }
  for (int hcol = threadIdx.x; hcol < H_; hcol += 256) {
    float acc = out[(size_t)t * H_ + hcol];
#pragma unroll
    for (int k = 0; k < KTOP; ++k)
      acc += w[k] * down_out[(size_t)rows[k] * H_ + hcol];
    out[(size_t)t * H_ + hcol] = acc;
  }
}

// --------------------------- workspace layout -------------------------------
constexpr size_t OFF_XB    = 0;                                    // T*H bf16
constexpr size_t OFF_QKV   = OFF_XB    + (size_t)T_ * H_ * 2;      // T*4096 f32
constexpr size_t OFF_QB    = OFF_QKV   + (size_t)T_ * QKVN * 4;    // T*NH*D bf16
constexpr size_t OFF_KB    = OFF_QB    + (size_t)T_ * NH_ * D_ * 2;
constexpr size_t OFF_VB    = OFF_KB    + (size_t)T_ * NKV_ * D_ * 2;
constexpr size_t OFF_ATT   = OFF_VB    + (size_t)T_ * NKV_ * D_ * 2;
constexpr size_t OFF_X2B   = OFF_ATT   + (size_t)T_ * NH_ * D_ * 2;
constexpr size_t OFF_LOG   = OFF_X2B   + (size_t)T_ * H_ * 2;
constexpr size_t OFF_CNT   = OFF_LOG   + (size_t)T_ * E_ * 4;
constexpr size_t OFF_BASE  = OFF_CNT   + 256;
constexpr size_t OFF_LIST  = OFF_BASE  + 256;
constexpr size_t OFF_SELE  = OFF_LIST  + (size_t)E_ * T_ * 4;
constexpr size_t OFF_SELP  = OFF_SELE  + (size_t)T_ * KTOP * 4;
constexpr size_t OFF_SELW  = OFF_SELP  + (size_t)T_ * KTOP * 4;
constexpr size_t OFF_ACTSH = OFF_SELW  + (size_t)T_ * KTOP * 4;    // T*SI bf16
constexpr size_t OFF_ACTEX = OFF_ACTSH + (size_t)T_ * SI_ * 2;     // 4T*I bf16
constexpr size_t OFF_DOWN  = OFF_ACTEX + (size_t)KTOP * T_ * I_ * 2; // 4T*H f32
// total ~164 MB

extern "C" void kernel_launch(void* const* d_in, const int* in_sizes, int n_in,
                              void* d_out, int out_size, void* d_ws, size_t ws_size,
                              hipStream_t stream) {
  const int*   positions = (const int*)d_in[0];
  const float* hs        = (const float*)d_in[1];
  const float* qkv_w     = (const float*)d_in[2];
  const float* o_w       = (const float*)d_in[3];
  const float* q_norm_w  = (const float*)d_in[4];
  const float* k_norm_w  = (const float*)d_in[5];
  const float* ln1_w     = (const float*)d_in[6];
  const float* ln2_w     = (const float*)d_in[7];
  const float* gate_w    = (const float*)d_in[8];
  const float* expert_b  = (const float*)d_in[9];
  const float* sh_gu_w   = (const float*)d_in[10];
  const float* sh_down_w = (const float*)d_in[11];
  const float* ex_gu_w   = (const float*)d_in[12];
  const float* ex_down_w = (const float*)d_in[13];
  float* out = (float*)d_out;

  char* ws = (char*)d_ws;
  __bf16* xb     = (__bf16*)(ws + OFF_XB);
  float*  qkv    = (float*)(ws + OFF_QKV);
  __bf16* qb     = (__bf16*)(ws + OFF_QB);
  __bf16* kbuf   = (__bf16*)(ws + OFF_KB);
  __bf16* vbuf   = (__bf16*)(ws + OFF_VB);
  __bf16* attnb  = (__bf16*)(ws + OFF_ATT);
  __bf16* x2b    = (__bf16*)(ws + OFF_X2B);
  float*  logits = (float*)(ws + OFF_LOG);
  int*    counts = (int*)(ws + OFF_CNT);
  int*    basev  = (int*)(ws + OFF_BASE);
  int*    tlist  = (int*)(ws + OFF_LIST);
  int*    sel_e  = (int*)(ws + OFF_SELE);
  int*    sel_p  = (int*)(ws + OFF_SELP);
  float*  sel_w  = (float*)(ws + OFF_SELW);
  __bf16* act_sh = (__bf16*)(ws + OFF_ACTSH);
  __bf16* act_ex = (__bf16*)(ws + OFF_ACTEX);
  float*  downo  = (float*)(ws + OFF_DOWN);

  // 1) x = rmsnorm(hs, ln1) -> bf16
  rmsnorm_to_bf16_kernel<<<T_, 256, 0, stream>>>(hs, ln1_w, xb, H_);
  // 2) qkv = x @ qkv_w
  gemm_bf16_kernel<0><<<dim3(T_ / 128, QKVN / 128), 256, 0, stream>>>(
      xb, qkv_w, qkv, nullptr, T_, QKVN, H_);
  // 3) qk-norm + rope + pack q/k/v bf16
  qk_rope_kernel<<<T_, 128, 0, stream>>>(qkv, positions, q_norm_w, k_norm_w,
                                         qb, kbuf, vbuf);
  // 4) flash attention
  attention_kernel<<<dim3(NH_, T_ / 128), 256, ATT_LDS, stream>>>(qb, kbuf, vbuf, attnb);
  // 5) h = hs + attn @ o_w   (writes d_out)
  gemm_bf16_kernel<1><<<dim3(T_ / 128, H_ / 128), 256, 0, stream>>>(
      attnb, o_w, out, hs, T_, H_, NH_ * D_);
  // 6) x2 = rmsnorm(h, ln2) -> bf16
  rmsnorm_to_bf16_kernel<<<T_, 256, 0, stream>>>(out, ln2_w, x2b, H_);
  // 7) router
  router_logits_kernel<<<T_ / 16, 256, 0, stream>>>(x2b, gate_w, logits);
  zero_counts_kernel<<<1, 64, 0, stream>>>(counts);
  topk_route_kernel<<<T_ / 256, 256, 0, stream>>>(logits, expert_b, counts,
                                                  sel_e, sel_p, sel_w, tlist);
  prefix_kernel<<<1, 32, 0, stream>>>(counts, basev);
  // 8) shared MLP: act = silu(g)*u ; out += act @ down
  glu_gemm_kernel<false><<<dim3(T_ / 128, SI_ / 64, 1), 256, 0, stream>>>(
      x2b, sh_gu_w, act_sh, H_, SI_, nullptr, nullptr, nullptr, T_);
  gemm_bf16_kernel<2><<<dim3(T_ / 128, H_ / 128), 256, 0, stream>>>(
      act_sh, sh_down_w, out, nullptr, T_, H_, SI_);
  // 9) routed experts: gather-GLU -> down -> combine
  glu_gemm_kernel<true><<<dim3(T_ / 128, I_ / 64, E_), 256, 0, stream>>>(
      x2b, ex_gu_w, act_ex, H_, I_, tlist, counts, basev, T_);
  expert_down_gemm_kernel<<<dim3(T_ / 128, H_ / 128, E_), 256, 0, stream>>>(
      act_ex, ex_down_w, downo, counts, basev);
  combine_kernel<<<T_, 256, 0, stream>>>(sel_e, sel_p, sel_w, basev, downo, out);
}
